// BiologicalBrain_36120674959575
// MI455X (gfx1250) — matrix operand order repaired and checked by reading, not compiled, not run on hardware
//
#include <hip/hip_runtime.h>
#include <hip/hip_bf16.h>
#include <cmath>

typedef __attribute__((ext_vector_type(16))) __bf16 v16bf;
typedef __attribute__((ext_vector_type(8)))  float  v8f;
typedef __attribute__((ext_vector_type(4)))  float  fx4;   // true clang vector: OK for nontemporal builtins

union BFrag {
  v16bf v;
  unsigned short u[16];
  unsigned int   w[8];
  uint4 q[2];
};

// Pack two f32 -> packed bf16x2 (round-half-up) with one v_perm_b32.
// perm(src0=ub, src1=ua, 0x07060302): dst = { ub[31:16], ua[31:16] }.
__device__ __forceinline__ unsigned int pack_bf16(float a, float b) {
  unsigned int ua = __builtin_bit_cast(unsigned int, a) + 0x8000u;
  unsigned int ub = __builtin_bit_cast(unsigned int, b) + 0x8000u;
  return __builtin_amdgcn_perm(ub, ua, 0x07060302u);
}

// clamp to [0,1] in one VALU op (v_med3_f32)
__device__ __forceinline__ float clip01(float m) {
  return __builtin_amdgcn_fmed3f(m, 0.f, 1.f);
}

// Build a 16x32 bf16 A-fragment for one lane from a row-major f32 row.
// p must point at  rowbase + kk + hi*16  (hi = lane>>4).  Layout:
// low lanes: halves 0..15 -> K{0..15}; high lanes +16.
__device__ __forceinline__ v16bf load_a_frag(const float* __restrict__ p, float scale) {
  fx4 x0 = *(const fx4*)(p);
  fx4 x1 = *(const fx4*)(p + 4);
  fx4 x2 = *(const fx4*)(p + 8);
  fx4 x3 = *(const fx4*)(p + 12);
  BFrag a;
  a.w[0] = pack_bf16(x0[0] * scale, x0[1] * scale);
  a.w[1] = pack_bf16(x0[2] * scale, x0[3] * scale);
  a.w[2] = pack_bf16(x1[0] * scale, x1[1] * scale);
  a.w[3] = pack_bf16(x1[2] * scale, x1[3] * scale);
  a.w[4] = pack_bf16(x2[0] * scale, x2[1] * scale);
  a.w[5] = pack_bf16(x2[2] * scale, x2[3] * scale);
  a.w[6] = pack_bf16(x3[0] * scale, x3[1] * scale);
  a.w[7] = pack_bf16(x3[2] * scale, x3[3] * scale);
  return a.v;
}

// ---------------------------------------------------------------------------
// gate[i] = (mean |Z[:, i, :]|) > 0.02 ? 1 : 0     (i = 0..3)
// ---------------------------------------------------------------------------
__global__ void gate_kernel(const float* __restrict__ Z, float* __restrict__ gate) {
  __shared__ float sdata[256];
  const int i = blockIdx.x;
  float s = 0.f;
  for (int idx = threadIdx.x; idx < 64 * 2048; idx += 256) {
    int b = idx >> 11, a = idx & 2047;
    s += fabsf(Z[((size_t)(b * 4 + i) << 11) + a]);
  }
  sdata[threadIdx.x] = s;
  __syncthreads();
  for (int off = 128; off > 0; off >>= 1) {
    if (threadIdx.x < off) sdata[threadIdx.x] += sdata[threadIdx.x + off];
    __syncthreads();
  }
  if (threadIdx.x == 0)
    gate[i] = (sdata[0] * (1.f / 131072.f) > 0.02f) ? 1.f : 0.f;
}

// ---------------------------------------------------------------------------
// stim[64, 8192] = x[64,3072] @ receptors_w[8192,3072]^T + receptors_b
// Block: 128 threads = 4 wave32 covering M=64; one 16-col tile per block.
// K-step 128 = 4 WMMAs per barrier; LDS ping-pong -> 1 barrier/iteration.
// ---------------------------------------------------------------------------
__global__ void stim_kernel(const float* __restrict__ x, const float* __restrict__ rw,
                            const float* __restrict__ rb, float* __restrict__ stim) {
  __shared__ __align__(16) unsigned short ldsB[2][2048]; // ping-pong, each: 4 x (32K x 16N)
  const int K = 3072;
  const int n0 = blockIdx.x * 16;
  const int t = threadIdx.x;
  const int lane = t & 31, wave = t >> 5;
  const int lo = lane & 15, hi = lane >> 4;
  const int m0 = wave * 16;
  // cooperative B loader: thread -> (column, 16 contiguous K)
  const int col_w = t >> 3;            // 0..15
  const int kseg  = (t & 7) * 16;      // 0,16,...,112
  const int sub   = kseg >> 5;         // sub-tile 0..3
  const int lane_w = col_w + ((kseg & 16) ? 16 : 0);
  const int doff = sub * 512 + lane_w * 16;

  v8f acc = {};
  int pp = 0;
  for (int kk = 0; kk < K; kk += 128, pp ^= 1) {
    const float* wp = rw + (size_t)(n0 + col_w) * K + kk + kseg;
    fx4 w0 = __builtin_nontemporal_load((const fx4*)(wp));
    fx4 w1 = __builtin_nontemporal_load((const fx4*)(wp + 4));
    fx4 w2 = __builtin_nontemporal_load((const fx4*)(wp + 8));
    fx4 w3 = __builtin_nontemporal_load((const fx4*)(wp + 12));
    uint4 s0, s1;
    s0.x = pack_bf16(w0[0], w0[1]);  s0.y = pack_bf16(w0[2], w0[3]);
    s0.z = pack_bf16(w1[0], w1[1]);  s0.w = pack_bf16(w1[2], w1[3]);
    s1.x = pack_bf16(w2[0], w2[1]);  s1.y = pack_bf16(w2[2], w2[3]);
    s1.z = pack_bf16(w3[0], w3[1]);  s1.w = pack_bf16(w3[2], w3[3]);
    uint4* dq = (uint4*)&ldsB[pp][doff];
    dq[0] = s0;
    dq[1] = s1;
    __syncthreads();
#pragma unroll
    for (int s = 0; s < 4; ++s) {
      BFrag bf;
      const uint4* lp = (const uint4*)&ldsB[pp][s * 512 + lane * 16];
      bf.q[0] = lp[0]; bf.q[1] = lp[1];
      v16bf av = load_a_frag(x + (size_t)(m0 + lo) * K + kk + s * 32 + hi * 16, 1.0f);
      acc = __builtin_amdgcn_wmma_f32_16x16x32_bf16(false, av, false, bf.v,
                                                    (short)0, acc, false, false);
    }
    // no trailing barrier: next iteration writes the other LDS buffer
  }
  const int n = n0 + lo;
  const float bn = rb[n];
#pragma unroll
  for (int r = 0; r < 8; ++r) {
    int m = m0 + hi * 8 + r;
    stim[(size_t)m * 8192 + n] = acc[r] + bn;   // RETINA_STRENGTH == 1.0
  }
}

// ---------------------------------------------------------------------------
// Fused main pass: Z_next = (Z*gate) @ (W*clip(mask))  [M=64,N=8192,K=8192]
//   + bias*gate + stim - (0.8*F + 0.4*Z), tanh, scatter by area_idx.
// W/mask streamed once with NT hints; W_eff formed in registers.
// ---------------------------------------------------------------------------
__global__ void brain_kernel(const float* __restrict__ Z, const float* __restrict__ F,
                             const float* __restrict__ W, const float* __restrict__ mask,
                             const float* __restrict__ bias, const float* __restrict__ gate,
                             const float* __restrict__ stim, const long long* __restrict__ aidx,
                             float* __restrict__ zflat) {
  __shared__ __align__(16) unsigned short ldsB[2][2048];
  const int K = 8192;
  const int n0 = blockIdx.x * 16;
  const int o  = n0 >> 11;          // output area (tile never crosses areas)
  const int u0 = n0 & 2047;
  const int t = threadIdx.x;
  const int lane = t & 31, wave = t >> 5;
  const int lo = lane & 15, hi = lane >> 4;
  const int m0 = wave * 16;
  const int col_w = t >> 3;
  const int kseg  = (t & 7) * 16;
  const int sub   = kseg >> 5;
  const int lane_w = col_w + ((kseg & 16) ? 16 : 0);
  const int doff = sub * 512 + lane_w * 16;

  v8f acc = {};
  int pp = 0;
  for (int kk = 0; kk < K; kk += 128, pp ^= 1) {
    const int i = kk >> 11;          // input area; constant across the 128-step
    const float gi = gate[i];
    size_t wb = ((size_t)(o * 4 + i) * 2048 + (size_t)(u0 + col_w)) * 2048
              + (kk & 2047) + kseg;
    const float* wp = W + wb;
    const float* mp = mask + wb;
    fx4 w0 = __builtin_nontemporal_load((const fx4*)(wp));
    fx4 w1 = __builtin_nontemporal_load((const fx4*)(wp + 4));
    fx4 w2 = __builtin_nontemporal_load((const fx4*)(wp + 8));
    fx4 w3 = __builtin_nontemporal_load((const fx4*)(wp + 12));
    fx4 m0v = __builtin_nontemporal_load((const fx4*)(mp));
    fx4 m1v = __builtin_nontemporal_load((const fx4*)(mp + 4));
    fx4 m2v = __builtin_nontemporal_load((const fx4*)(mp + 8));
    fx4 m3v = __builtin_nontemporal_load((const fx4*)(mp + 12));
    uint4 s0, s1;
    s0.x = pack_bf16(w0[0]*clip01(m0v[0]), w0[1]*clip01(m0v[1]));
    s0.y = pack_bf16(w0[2]*clip01(m0v[2]), w0[3]*clip01(m0v[3]));
    s0.z = pack_bf16(w1[0]*clip01(m1v[0]), w1[1]*clip01(m1v[1]));
    s0.w = pack_bf16(w1[2]*clip01(m1v[2]), w1[3]*clip01(m1v[3]));
    s1.x = pack_bf16(w2[0]*clip01(m2v[0]), w2[1]*clip01(m2v[1]));
    s1.y = pack_bf16(w2[2]*clip01(m2v[2]), w2[3]*clip01(m2v[3]));
    s1.z = pack_bf16(w3[0]*clip01(m3v[0]), w3[1]*clip01(m3v[1]));
    s1.w = pack_bf16(w3[2]*clip01(m3v[2]), w3[3]*clip01(m3v[3]));
    uint4* dq = (uint4*)&ldsB[pp][doff];
    dq[0] = s0;
    dq[1] = s1;
    __syncthreads();
#pragma unroll
    for (int s = 0; s < 4; ++s) {
      BFrag bf;
      const uint4* lp = (const uint4*)&ldsB[pp][s * 512 + lane * 16];
      bf.q[0] = lp[0]; bf.q[1] = lp[1];
      v16bf av = load_a_frag(Z + (size_t)(m0 + lo) * K + kk + s * 32 + hi * 16, gi);
      acc = __builtin_amdgcn_wmma_f32_16x16x32_bf16(false, av, false, bf.v,
                                                    (short)0, acc, false, false);
    }
    // no trailing barrier: ping-pong LDS
  }
  // --- fused epilogue ---
  const int n = n0 + lo;                     // n = o*2048 + u
  const float go = gate[o];
  const float bd = bias[n] * go;             // bias_diag flat index == n
  const long long dst = aidx[n];             // identity in reference, honored anyway
#pragma unroll
  for (int r = 0; r < 8; ++r) {
    int m = m0 + hi * 8 + r;
    size_t idx = (size_t)m * 8192 + n;
    float fnew = F[idx] * 0.8f + Z[idx] * 0.4f;
    float v = acc[r] + bd + stim[idx] - fnew;
    zflat[(size_t)m * 8192 + (size_t)dst] = tanhf(v);
  }
}

// ---------------------------------------------------------------------------
// raw[64,11] = zflat @ out_w^T + out_b; sigmoid on column 10.
// ---------------------------------------------------------------------------
__global__ void out_kernel(const float* __restrict__ zflat, const float* __restrict__ ow,
                           const float* __restrict__ ob, float* __restrict__ out) {
  __shared__ float sdata[256];
  const int b = blockIdx.x;
  for (int j = 0; j < 11; ++j) {
    float s = 0.f;
    for (int n = threadIdx.x; n < 8192; n += 256)
      s += zflat[(size_t)b * 8192 + n] * ow[(size_t)j * 8192 + n];
    sdata[threadIdx.x] = s;
    __syncthreads();
    for (int off = 128; off > 0; off >>= 1) {
      if (threadIdx.x < off) sdata[threadIdx.x] += sdata[threadIdx.x + off];
      __syncthreads();
    }
    if (threadIdx.x == 0) {
      float raw = sdata[0] + ob[j];
      out[b * 11 + j] = (j == 10) ? (1.f / (1.f + expf(-raw))) : raw;
    }
    __syncthreads();
  }
}

extern "C" void kernel_launch(void* const* d_in, const int* in_sizes, int n_in,
                              void* d_out, int out_size, void* d_ws, size_t ws_size,
                              hipStream_t stream) {
  const float*     x  = (const float*)d_in[0];
  const float*     Z  = (const float*)d_in[1];
  const float*     F  = (const float*)d_in[2];
  const float*     rw = (const float*)d_in[3];
  const float*     rb = (const float*)d_in[4];
  const float*     W  = (const float*)d_in[5];
  const float*     mk = (const float*)d_in[6];
  const float*     bd = (const float*)d_in[7];
  const float*     ow = (const float*)d_in[8];
  const float*     ob = (const float*)d_in[9];
  const long long* ai = (const long long*)d_in[10];

  float* gate  = (float*)d_ws;                                        // 4 floats
  float* stim  = (float*)((char*)d_ws + 256);                         // 64*8192 f32 = 2 MB
  float* zflat = (float*)((char*)d_ws + 256 + (size_t)64 * 8192 * 4); // 2 MB

  gate_kernel <<<4,   256, 0, stream>>>(Z, gate);
  stim_kernel <<<512, 128, 0, stream>>>(x, rw, rb, stim);
  brain_kernel<<<512, 128, 0, stream>>>(Z, F, W, mk, bd, gate, stim, ai, zflat);
  out_kernel  <<<64,  256, 0, stream>>>(zflat, ow, ob, (float*)d_out);
}
